// KroneckerProduct_61942018342941
// MI455X (gfx1250) — matrix operand order repaired and checked by reading, not compiled, not run on hardware
//
#include <hip/hip_runtime.h>
#include <hip/hip_bf16.h>

// ---------------- problem constants ----------------
#define BTOK   (8 * 2048)     // B*T tokens
#define INF    1024           // IN_F = OUT_F
#define P1     528            // triu params per factor
#define NP     1056           // 2*P1
#define ND     64             // M1+M2 diag outputs
#define NTOT   1120           // NP + ND = 70*16
#define M32    32
#define BM     256            // GEMM block tile M
#define BN     32             // GEMM block tile N
#define KT     (INF / 32)     // K tiles of 32

typedef __attribute__((ext_vector_type(16))) __bf16 v16bf;
typedef __attribute__((ext_vector_type(8)))  __bf16 v8bf;
typedef __attribute__((ext_vector_type(4)))  __bf16 v4bf;
typedef __attribute__((ext_vector_type(8)))  float  v8f;

#if defined(__has_builtin)
#  if __has_builtin(__builtin_amdgcn_tensor_load_to_lds) && \
      __has_builtin(__builtin_amdgcn_s_wait_tensorcnt)
#    define HAVE_TDM 1
#  endif
#endif
#ifndef HAVE_TDM
#  define HAVE_TDM 0
#endif

#if HAVE_TDM
typedef __attribute__((ext_vector_type(4))) unsigned int u32x4;
typedef __attribute__((ext_vector_type(8))) int          i32x8;
typedef __attribute__((ext_vector_type(4))) int          i32x4;
#endif

#define SHUF16(lo, hi) \
  __builtin_shufflevector(lo, hi, 0, 1, 2, 3, 4, 5, 6, 7, 8, 9, 10, 11, 12, 13, 14, 15)

// ---------------------------------------------------------------------------
// Kernel 1: convert param_arg (f32) -> bf16 (halves HBM re-reads in the GEMM)
// ---------------------------------------------------------------------------
__global__ __launch_bounds__(256) void convert_p_kernel(
    const float* __restrict__ src, __bf16* __restrict__ dst, int n) {
  int i = (blockIdx.x * 256 + threadIdx.x) * 8;
  if (i + 8 <= n) {
#pragma unroll
    for (int k = 0; k < 8; ++k) dst[i + k] = (__bf16)src[i + k];
  }
}

// ---------------------------------------------------------------------------
// Kernel 2: combined transposed bf16 weights Wt[k][n], n in [0,1120)
// ---------------------------------------------------------------------------
__global__ __launch_bounds__(256) void build_wt_kernel(
    const float* __restrict__ Wuv, const float* __restrict__ Wd,
    __bf16* __restrict__ Wt) {
  int idx = blockIdx.x * 256 + threadIdx.x;   // over 1024*1120
  if (idx >= INF * NTOT) return;
  int k = idx / NTOT;
  int n = idx - k * NTOT;
  float v = (n < NP) ? Wuv[(size_t)n * INF + k]
                     : Wd[(size_t)(n - NP) * INF + k];
  Wt[idx] = (__bf16)v;
}

// ---------------------------------------------------------------------------
// Kernel 3: C[BT x 1120] = Pbf[BT x 1024] @ Wt[1024 x 1120], f32 accumulate.
// Block 256 thr (8 waves), block tile 256(M) x 32(N), K chunks of 32,
// double-buffered LDS. A tile staged by the Tensor Data Mover (wave 0 issues
// a 2D D# each K-step, overlapped with compute, s_wait_tensorcnt to close);
// B tile staged transposed so every fragment is two ds_load_b128.
// Wave w -> 32x32 C strip: 2 A frags x 2 B frags -> 4 v_wmma_f32_16x16x32_bf16.
// ---------------------------------------------------------------------------
__global__ __launch_bounds__(256) void gemm_kernel(
    const __bf16* __restrict__ P, const __bf16* __restrict__ Wt,
    float* __restrict__ C) {
  __shared__ __align__(16) __bf16 As[2][BM][32];
  __shared__ __align__(16) __bf16 BsT[2][BN][32];   // BsT[n][k]

  const int tid  = threadIdx.x;
  const int wave = tid >> 5;
  const int lane = tid & 31;
  const int m0   = blockIdx.x * BM;
  const int n0   = blockIdx.y * BN;
  const int m    = lane & 15;   // row within 16x16 tile
  const int h    = lane >> 4;   // lane-half selector
  const int n    = lane & 15;   // col within tile

  // B loader indices (256 thr x 4 elems)
  const int bk = tid >> 3;          // 0..31
  const int bn = (tid & 7) * 4;     // 0..28

#if HAVE_TDM
  // TDM descriptor group 1 (loop-invariant): data_size=2B, tensor_dim0=1024,
  // tensor_dim1=16384, tile 32x256, dim0_stride=1024.
  i32x8 g1;
  g1[0] = (int)(1u << 16);        // data_size code 1 (2 bytes)
  g1[1] = (int)(1024u << 16);     // tensor_dim0[15:0] @ bits 63:48
  g1[2] = (int)(16384u << 16);    // tensor_dim0 hi=0 | tensor_dim1[15:0]
  g1[3] = (int)(32u << 16);       // tensor_dim1 hi=0 | tile_dim0=32
  g1[4] = (int)BM;                // tile_dim1=256, tile_dim2=0
  g1[5] = (int)1024;              // tensor_dim0_stride[31:0]
  g1[6] = 0;
  g1[7] = 0;
  i32x4 gz4 = {0, 0, 0, 0};
  i32x8 gz8 = {0, 0, 0, 0, 0, 0, 0, 0};   // groups 2/3 unused (2D tile)
#endif

  auto stageA = [&](int kt, int buf) {
#if HAVE_TDM
    if (wave == 0) {
      unsigned long long ga =
          (unsigned long long)(uintptr_t)(P + (size_t)m0 * INF + kt * 32);
      u32x4 g0;
      g0[0] = 1u;                                       // count=1, user desc
      g0[1] = (unsigned)(uintptr_t)&As[buf][0][0];      // LDS byte address
      g0[2] = (unsigned)ga;                             // global addr lo
      g0[3] = (unsigned)((ga >> 32) & 0x01FFFFFFu) | (2u << 30);  // hi | type=2
      __builtin_amdgcn_tensor_load_to_lds(g0, g1, gz4, gz4, gz8, 0);
    }
#else
    const __bf16* gA = P + (size_t)(m0 + tid) * INF + kt * 32;
    __builtin_prefetch(gA + 32, 0, 1);                  // global_prefetch_b8
    v8bf* dst = (v8bf*)&As[buf][tid][0];
#pragma unroll
    for (int c = 0; c < 4; ++c) dst[c] = *(const v8bf*)(gA + c * 8);
#endif
  };

  auto stageB = [&](int kt, int buf) {
    const __bf16* gB = Wt + (size_t)(kt * 32 + bk) * NTOT + n0 + bn;
    __builtin_prefetch(gB + (size_t)32 * NTOT, 0, 1);   // next K tile
    v4bf w = *(const v4bf*)gB;
    BsT[buf][bn + 0][bk] = w[0];
    BsT[buf][bn + 1][bk] = w[1];
    BsT[buf][bn + 2][bk] = w[2];
    BsT[buf][bn + 3][bk] = w[3];
  };

  v8f acc[2][2] = {{{}, {}}, {{}, {}}};

  stageA(0, 0);
  stageB(0, 0);
#if HAVE_TDM
  if (wave == 0) __builtin_amdgcn_s_wait_tensorcnt(0);
#endif
  __syncthreads();

  for (int kt = 0; kt < KT; ++kt) {
    const int cur = kt & 1;
    if (kt + 1 < KT) {          // prefetch next tiles into the other buffer
      stageA(kt + 1, cur ^ 1);
      stageB(kt + 1, cur ^ 1);
    }

    // fragments: contiguous 16B LDS vector loads, packed per gfx1250 layouts
    v16bf av[2], bv[2];
#pragma unroll
    for (int mi = 0; mi < 2; ++mi) {
      const int row = wave * 32 + mi * 16 + m;
      v8bf lo = *(const v8bf*)&As[cur][row][h * 8];
      v8bf hi = *(const v8bf*)&As[cur][row][16 + h * 8];
      av[mi] = SHUF16(lo, hi);
    }
#pragma unroll
    for (int ni = 0; ni < 2; ++ni) {
      const int col = ni * 16 + n;
      v8bf lo = *(const v8bf*)&BsT[cur][col][h * 8];
      v8bf hi = *(const v8bf*)&BsT[cur][col][16 + h * 8];
      bv[ni] = SHUF16(lo, hi);
    }
#pragma unroll
    for (int mi = 0; mi < 2; ++mi)
#pragma unroll
      for (int ni = 0; ni < 2; ++ni)
        acc[mi][ni] = __builtin_amdgcn_wmma_f32_16x16x32_bf16(
            false, av[mi], false, bv[ni], (short)0, acc[mi][ni], false, false);

#if HAVE_TDM
    if (wave == 0 && kt + 1 < KT) __builtin_amdgcn_s_wait_tensorcnt(0);
#endif
    __syncthreads();
  }

  // store C tiles (f32 C layout: VGPR r -> row h*8+r, col = lane&15)
#pragma unroll
  for (int mi = 0; mi < 2; ++mi)
#pragma unroll
    for (int ni = 0; ni < 2; ++ni)
#pragma unroll
      for (int r = 0; r < 8; ++r) {
        int row = m0 + wave * 32 + mi * 16 + h * 8 + r;
        C[(size_t)row * NTOT + n0 + ni * 16 + n] = acc[mi][ni][r];
      }
}

// ---------------------------------------------------------------------------
// Kernel 4: fused per-token pipeline, 1 block (128 thr / 4 waves) per token.
// bf16 LDS mirrors are laid out so every WMMA fragment is two ds_load_b128.
// ---------------------------------------------------------------------------
__global__ __launch_bounds__(128) void token_kernel(
    const float* __restrict__ params,   // BT x 1120  ([u|v|diag logits])
    const float* __restrict__ inp,      // BT x 1024
    const float* __restrict__ b_diag,   // 64
    float* __restrict__ out) {          // BT x 1024
  __shared__ float Gu[M32][M32 + 1];
  __shared__ float Gv[M32][M32 + 1];
  __shared__ __align__(16) __bf16 Mb0[M32][M32];  // Tri_u | X (row-major)
  __shared__ __align__(16) __bf16 Mb1[M32][M32];  // Tri_v | Um^T
  __shared__ __align__(16) __bf16 Mb2[M32][M32];  // V_mat (row-major)
  __shared__ __align__(16) __bf16 Mb3[M32][M32];  // T^T
  __shared__ float dg[ND];
  __shared__ float rsU[M32], rsV[M32];

  const int tid  = threadIdx.x;
  const int wave = tid >> 5;
  const int lane = tid & 31;
  const size_t tok = blockIdx.x;
  const float* gp = params + tok * NTOT;
  const float* gx = inp    + tok * INF;

  const int mi = wave >> 1, ni = wave & 1;
  const int m = lane & 15, h = lane >> 4, n = lane & 15;

  auto fragA = [&](const __bf16 (*Mt)[M32], int row) -> v16bf {
    v8bf lo = *(const v8bf*)&Mt[row][h * 8];
    v8bf hi = *(const v8bf*)&Mt[row][16 + h * 8];
    return SHUF16(lo, hi);
  };
  auto fragB = [&](const __bf16 (*Mt)[M32], int row) -> v16bf {
    v8bf lo = *(const v8bf*)&Mt[row][h * 8];
    v8bf hi = *(const v8bf*)&Mt[row][16 + h * 8];
    return SHUF16(lo, hi);
  };

  // zero triangular staging + sigmoid(diag logits + bias)
  for (int i = tid; i < 1024; i += 128) {
    Gu[i >> 5][i & 31] = 0.f;
    Gv[i >> 5][i & 31] = 0.f;
  }
  if (tid < ND) dg[tid] = 1.f / (1.f + __expf(-(gp[NP + tid] + b_diag[tid])));
  __syncthreads();

  // scatter upper-triangular params
  for (int k = tid; k < P1; k += 128) {
    int i = 0, off = 0;
    while (k >= off + (M32 - i)) { off += M32 - i; ++i; }
    int j = i + (k - off);
    Gu[i][j] = gp[k];
    Gv[i][j] = gp[P1 + k];
  }
  __syncthreads();

  // bf16 mirrors of the triangular factors (row-major serves both A and B^T)
  for (int i = tid; i < 1024; i += 128) {
    Mb0[i >> 5][i & 31] = (__bf16)Gu[i >> 5][i & 31];
    Mb1[i >> 5][i & 31] = (__bf16)Gv[i >> 5][i & 31];
  }
  __syncthreads();

  // G = Tri @ Tri^T : B[k][n] = Tri[n][k] -> B fragment = Tri row n
  v8f cu = {}, cv = {};
  cu = __builtin_amdgcn_wmma_f32_16x16x32_bf16(
      false, fragA(Mb0, mi * 16 + m), false, fragB(Mb0, ni * 16 + n),
      (short)0, cu, false, false);
  cv = __builtin_amdgcn_wmma_f32_16x16x32_bf16(
      false, fragA(Mb1, mi * 16 + m), false, fragB(Mb1, ni * 16 + n),
      (short)0, cv, false, false);
  __syncthreads();
#pragma unroll
  for (int r = 0; r < 8; ++r) {
    Gu[mi * 16 + h * 8 + r][ni * 16 + n] = cu[r];
    Gv[mi * 16 + h * 8 + r][ni * 16 + n] = cv[r];
  }
  __syncthreads();

  // row-wise L2 normalize: x / max(||x||, 1e-5)
  if (tid < M32) {
    float ss = 0.f;
    for (int j = 0; j < M32; ++j) ss += Gu[tid][j] * Gu[tid][j];
    rsU[tid] = 1.f / fmaxf(sqrtf(ss), 1e-5f);
  } else if (tid < 2 * M32) {
    int r = tid - M32;
    float ss = 0.f;
    for (int j = 0; j < M32; ++j) ss += Gv[r][j] * Gv[r][j];
    rsV[r] = 1.f / fmaxf(sqrtf(ss), 1e-5f);
  }
  __syncthreads();
  for (int i = tid; i < 1024; i += 128) {
    Gu[i >> 5][i & 31] *= rsU[i >> 5];
    Gv[i >> 5][i & 31] *= rsV[i >> 5];
  }
  __syncthreads();

  // 16 power iterations, wave-resident: wave0 -> Gu, wave1 -> Gv
  if (tid < 64) {
    const bool isU = (tid < 32);
    float (*G)[M32 + 1] = isU ? Gu : Gv;
    float vv = 0.03125f;                 // 1/sqrt(OUT_F)
    float uu = vv;
#pragma unroll 1
    for (int it = 0; it < 16; ++it) {
      float x = (it & 1) ? uu : vv;
      float y = 0.f;
      for (int j = 0; j < M32; ++j) y += G[lane][j] * __shfl(x, j, 32);
      float ss = y * y;
#pragma unroll
      for (int d = 16; d; d >>= 1) ss += __shfl_xor(ss, d, 32);
      y *= 1.f / fmaxf(sqrtf(ss), 1e-12f);
      if (it & 1) vv = y; else uu = y;
    }
    float r = 0.f;                       // spec = u^T G v
    for (int j = 0; j < M32; ++j) r += G[lane][j] * __shfl(vv, j, 32);
    float sp = uu * r;
#pragma unroll
    for (int d = 16; d; d >>= 1) sp += __shfl_xor(sp, d, 32);
    float scale = dg[isU ? lane : (M32 + lane)] / (sp + 1e-5f);
    (isU ? rsU : rsV)[lane] = scale;     // fused spectral + diag row scale
  }
  __syncthreads();
  for (int i = tid; i < 1024; i += 128) {
    Gu[i >> 5][i & 31] *= rsU[i >> 5];   // Gu -> U_mat
    Gv[i >> 5][i & 31] *= rsV[i >> 5];   // Gv -> V_mat
  }
  __syncthreads();

  // bf16 mirrors for the application matmuls:
  //   Mb0 = X (row-major, X[j][i] = inp[i*32+j]); Mb1 = Um^T; Mb2 = V row-major
  for (int i = tid; i < 1024; i += 128) {
    int r = i >> 5, c = i & 31;
    Mb0[c][r] = (__bf16)gx[i];
    Mb1[c][r] = (__bf16)Gu[r][c];
    Mb2[r][c] = (__bf16)Gv[r][c];
  }
  __syncthreads();

  // T = X @ Um : B[k][n] = Um[k][n] = UmT[n][k] -> fragB of Mb1 row n
  v8f c1 = {};
  c1 = __builtin_amdgcn_wmma_f32_16x16x32_bf16(
      false, fragA(Mb0, mi * 16 + m), false, fragB(Mb1, ni * 16 + n),
      (short)0, c1, false, false);
#pragma unroll
  for (int r = 0; r < 8; ++r)                // store T transposed
    Mb3[ni * 16 + n][mi * 16 + h * 8 + r] = (__bf16)c1[r];
  __syncthreads();

  // out = V @ T : B[k][n] = T[k][n] = Mb3[n][k] -> fragB of Mb3 row n
  v8f c2 = {};
  c2 = __builtin_amdgcn_wmma_f32_16x16x32_bf16(
      false, fragA(Mb2, mi * 16 + m), false, fragB(Mb3, ni * 16 + n),
      (short)0, c2, false, false);
  float* go = out + tok * INF;
#pragma unroll
  for (int r = 0; r < 8; ++r) {
    int j  = mi * 16 + h * 8 + r;        // row of out (M2 index)
    int i2 = ni * 16 + n;                // col of out (M1 index)
    int f  = i2 * 32 + j;
    go[f] = gx[f] - 2.f * c2[r];
  }
}

// ---------------------------------------------------------------------------
extern "C" void kernel_launch(void* const* d_in, const int* in_sizes, int n_in,
                              void* d_out, int out_size, void* d_ws, size_t ws_size,
                              hipStream_t stream) {
  const float* param_arg = (const float*)d_in[0];   // (B,T,1024)
  const float* input_arg = (const float*)d_in[1];   // (B,T,1024)
  const float* W_uv      = (const float*)d_in[2];   // (1056,1024)
  const float* W_diag    = (const float*)d_in[3];   // (64,1024)
  const float* b_diag    = (const float*)d_in[4];   // (64,)
  float* out = (float*)d_out;

  // workspace layout
  __bf16* pbf = (__bf16*)d_ws;                              // BT*1024 bf16
  __bf16* wt  = pbf + (size_t)BTOK * INF;                   // 1024*1120 bf16
  float* paramsf = (float*)(wt + (size_t)INF * NTOT);       // BT*1120 f32

  const int np = BTOK * INF;
  convert_p_kernel<<<np / (256 * 8), 256, 0, stream>>>(param_arg, pbf, np);
  build_wt_kernel<<<(INF * NTOT + 255) / 256, 256, 0, stream>>>(W_uv, W_diag, wt);
  gemm_kernel<<<dim3(BTOK / BM, NTOT / BN), 256, 0, stream>>>(pbf, wt, paramsf);
  token_kernel<<<BTOK, 128, 0, stream>>>(paramsf, input_arg, b_diag, out);
}
